// RGCNNet_87600152969645
// MI455X (gfx1250) — matrix operand order repaired and checked by reading, not compiled, not run on hardware
//
#include <hip/hip_runtime.h>

// RGCN 2-layer forward for MI455X (gfx1250), wave32 + WMMA bf16.
// aggregate-first (mean is linear): scatter-mean features into bf16 M[N, R*din]
// via global_atomic_pk_add_bf16, then ONE wmma GEMM per layer:
//   out = M @ Wflat + x @ root + b   (root folded in as a second K phase)
// GEMM: one wave per 16-row tile; all NC col-tile accumulators + NC B fragments
// in registers; sched_barrier groups all loads ahead of the WMMA burst so the
// backend emits progressive s_wait_loadcnt (load/WMMA pipelining in-wave).

#define NN   50000
#define NE   625000
#define NR   8
#define DIN  128
#define DHID 128
#define DOUT 64

typedef __attribute__((ext_vector_type(16))) __bf16          v16bf;
typedef __attribute__((ext_vector_type(16))) unsigned short  v16u;
typedef __attribute__((ext_vector_type(8)))  unsigned short  v8u;
typedef __attribute__((ext_vector_type(8)))  float           v8f;

__device__ __forceinline__ unsigned short bf16_rne(float f) {
  unsigned int u = __float_as_uint(f);
  unsigned int r = u + 0x7FFFu + ((u >> 16) & 1u);   // round-to-nearest-even
  return (unsigned short)(r >> 16);
}

// CDNA5 packed bf16 atomic add (GLOBAL_ATOMIC_PK_ADD_BF16), no return.
__device__ __forceinline__ void atomic_pk_add_bf16(unsigned short* addr, unsigned int pk2) {
  asm volatile("global_atomic_pk_add_bf16 %0, %1, off" : : "v"(addr), "v"(pk2) : "memory");
}

__device__ __forceinline__ v8f wmma_bf16(v16bf a, v16bf b, v8f c) {
  return __builtin_amdgcn_wmma_f32_16x16x32_bf16(false, a, false, b, (short)0, c,
                                                 false, false);
}

// A-fragment (contiguous k order, B uses the same): lane half -> K group {h*16..+15}
__device__ __forceinline__ v16bf load_a_bf16(const unsigned short* p) {
  v8u p0 = *(const v8u*)(p);
  v8u p1 = *(const v8u*)(p + 8);
  v16u au = __builtin_shufflevector(p0, p1, 0, 1, 2, 3, 4, 5, 6, 7,
                                    8, 9, 10, 11, 12, 13, 14, 15);
  return __builtin_bit_cast(v16bf, au);
}

__device__ __forceinline__ v16bf load_a_f32(const float* p) {
  float4 f0 = *(const float4*)(p);
  float4 f1 = *(const float4*)(p + 4);
  float4 f2 = *(const float4*)(p + 8);
  float4 f3 = *(const float4*)(p + 12);
  v16u au;
  au[0]  = bf16_rne(f0.x); au[1]  = bf16_rne(f0.y);
  au[2]  = bf16_rne(f0.z); au[3]  = bf16_rne(f0.w);
  au[4]  = bf16_rne(f1.x); au[5]  = bf16_rne(f1.y);
  au[6]  = bf16_rne(f1.z); au[7]  = bf16_rne(f1.w);
  au[8]  = bf16_rne(f2.x); au[9]  = bf16_rne(f2.y);
  au[10] = bf16_rne(f2.z); au[11] = bf16_rne(f2.w);
  au[12] = bf16_rne(f3.x); au[13] = bf16_rne(f3.y);
  au[14] = bf16_rne(f3.z); au[15] = bf16_rne(f3.w);
  return __builtin_bit_cast(v16bf, au);
}

// ---------------- utility kernels ----------------

__global__ void zero_u128(uint4* __restrict__ p, long n4) {
  long i = (long)blockIdx.x * blockDim.x + threadIdx.x;
  long stride = (long)gridDim.x * blockDim.x;
  uint4 z = {0u, 0u, 0u, 0u};
  for (; i < n4; i += stride) p[i] = z;
}

__global__ void count_edges(const int* __restrict__ dst, const int* __restrict__ et,
                            unsigned int* __restrict__ cnt, int E) {
  int t = blockIdx.x * blockDim.x + threadIdx.x;
  if (t < E) atomicAdd(&cnt[(long)dst[t] * NR + et[t]], 1u);
}

// in-place: cnt (uint) -> 1/max(cnt,1) (float), same buffer
__global__ void make_inv(unsigned int* __restrict__ cnt, float* inv, long n) {
  long i = (long)blockIdx.x * blockDim.x + threadIdx.x;
  if (i < n) {
    unsigned int c = cnt[i];
    inv[i] = 1.0f / (float)(c ? c : 1u);
  }
}

// src [rows, cols] f32 row-major  ->  dst [cols, rows] bf16 row-major (i.e. B^T)
__global__ void xpose_bf16(const float* __restrict__ src, unsigned short* __restrict__ dst,
                           int rows, int cols) {
  int t = blockIdx.x * blockDim.x + threadIdx.x;
  int total = rows * cols;
  if (t < total) {
    int k = t / cols;
    int j = t - k * cols;
    dst[(long)j * rows + k] = bf16_rne(src[t]);
  }
}

// one wave per edge: M[(dst*R+et)*128 + f] += bf16(x[src*128 + f] * inv[seg])
__global__ void scatter_mean_pk(const float* __restrict__ feat, const int* __restrict__ src,
                                const int* __restrict__ dst, const int* __restrict__ et,
                                const float* __restrict__ inv, unsigned short* __restrict__ M,
                                int E) {
  long t = (long)blockIdx.x * blockDim.x + threadIdx.x;
  int e = (int)(t >> 5);
  if (e >= E) return;
  int lo = (int)(t & 31);                     // lane within edge-wave
  int s = src[e], d = dst[e], r = et[e];      // uniform per wave -> scalar loads
  long seg = (long)d * NR + r;
  float w = inv[seg];
  float4 v = *(const float4*)(feat + (long)s * 128 + lo * 4);
  unsigned short* mp = M + seg * 128 + lo * 4;
  unsigned int p0 = (unsigned int)bf16_rne(v.x * w) | ((unsigned int)bf16_rne(v.y * w) << 16);
  unsigned int p1 = (unsigned int)bf16_rne(v.z * w) | ((unsigned int)bf16_rne(v.w * w) << 16);
  atomic_pk_add_bf16(mp, p0);
  atomic_pk_add_bf16(mp + 2, p1);
}

// ---------------- WMMA GEMM ----------------
// out[row0:16, 0:NC*16] = A1 @ Bt1^T + A2 @ Bt2^T + bias (opt relu)
// A1 is bf16 [N,K1]; A2 is f32 [N,K2]; Bt* are bf16 [dout, K*] (transposed).
// grid = N/16, block = 32 (one wave per row tile, EXEC all ones).
template <int NC, int K1, int K2>
__global__ void __launch_bounds__(32, 1)
rgcn_gemm_wmma(const unsigned short* __restrict__ A1,
               const float* __restrict__ A2,
               const unsigned short* __restrict__ Bt1,
               const unsigned short* __restrict__ Bt2,
               const float* __restrict__ bias,
               float* __restrict__ out, int relu) {
  constexpr int dout = NC * 16;
  int lane = threadIdx.x & 31;
  int nm   = lane & 15;
  int half = lane >> 4;
  int row0 = blockIdx.x << 4;

  v8f acc[NC];
#pragma unroll
  for (int c = 0; c < NC; ++c) acc[c] = (v8f){};

  // ---- phase 1: A1 (bf16) x Bt1, K = K1 ----
  {
    const unsigned short* ar = A1 + (long)(row0 + nm) * K1 + half * 16;
    const unsigned short* br = Bt1 + (long)nm * K1 + half * 16;
#pragma unroll 1
    for (int k = 0; k < K1; k += 32) {
      v16bf a = load_a_bf16(ar + k);
      v16bf b[NC];
#pragma unroll
      for (int c = 0; c < NC; ++c) b[c] = *(const v16bf*)(br + c * (16 * K1) + k);
      __builtin_amdgcn_sched_barrier(0);   // issue all loads before the WMMA burst
#pragma unroll
      for (int c = 0; c < NC; ++c) acc[c] = wmma_bf16(a, b[c], acc[c]);
    }
  }

  // ---- phase 2: A2 (f32 -> bf16) x Bt2, K = K2 ----
  {
    const float* ar = A2 + (long)(row0 + nm) * K2 + half * 16;
    const unsigned short* br = Bt2 + (long)nm * K2 + half * 16;
#pragma unroll 1
    for (int k = 0; k < K2; k += 32) {
      v16bf a = load_a_f32(ar + k);
      v16bf b[NC];
#pragma unroll
      for (int c = 0; c < NC; ++c) b[c] = *(const v16bf*)(br + c * (16 * K2) + k);
      __builtin_amdgcn_sched_barrier(0);
#pragma unroll
      for (int c = 0; c < NC; ++c) acc[c] = wmma_bf16(a, b[c], acc[c]);
    }
  }

  // ---- epilogue: bias, relu, store ----
#pragma unroll
  for (int c = 0; c < NC; ++c) {
    int col = c * 16 + nm;
    float bv = bias[col];
#pragma unroll
    for (int v = 0; v < 8; ++v) {
      int row = row0 + half * 8 + v;         // C/D layout: VGPR v -> M = v + half*8
      float val = acc[c][v] + bv;
      if (relu) val = fmaxf(val, 0.0f);
      out[(long)row * dout + col] = val;
    }
  }
}

// ---------------- launcher ----------------

extern "C" void kernel_launch(void* const* d_in, const int* in_sizes, int n_in,
                              void* d_out, int out_size, void* d_ws, size_t ws_size,
                              hipStream_t stream) {
  const float* x     = (const float*)d_in[0];
  const int*   ei    = (const int*)  d_in[1];
  const int*   etype = (const int*)  d_in[2];
  const float* W1    = (const float*)d_in[3];
  const float* root1 = (const float*)d_in[4];
  const float* b1    = (const float*)d_in[5];
  const float* W2    = (const float*)d_in[6];
  const float* root2 = (const float*)d_in[7];
  const float* b2    = (const float*)d_in[8];
  float*       out   = (float*)d_out;
  const int* src = ei;
  const int* dst = ei + NE;

  char* ws = (char*)d_ws;
  // workspace layout (all 256B aligned)
  unsigned int*   cnt = (unsigned int*)ws;                        // N*R u32 (1.6 MB)
  float*          inv = (float*)ws;                               // aliased in-place
  unsigned short* M   = (unsigned short*)(ws + (size_t)2097152);  // N*1024 bf16 (102.4 MB)
  float*          h   = (float*)(ws + (size_t)104497152);         // N*128 f32 (25.6 MB)
  unsigned short* W1t = (unsigned short*)(ws + (size_t)130097152);// [128,1024] bf16
  unsigned short* r1t = (unsigned short*)(ws + (size_t)130359296);// [128,128]  bf16
  unsigned short* W2t = (unsigned short*)(ws + (size_t)130392064);// [64,1024]  bf16
  unsigned short* r2t = (unsigned short*)(ws + (size_t)130523136);// [64,128]   bf16

  const long nseg = (long)NN * NR;               // 400,000
  const long nM4  = (long)NN * 1024 * 2 / 16;    // M bytes / 16 = 6.4M uint4

  // counts (shared by both layers) and weight repack
  zero_u128<<<dim3(512),  dim3(256), 0, stream>>>((uint4*)cnt, nseg / 4);
  zero_u128<<<dim3(4096), dim3(256), 0, stream>>>((uint4*)M, nM4);
  count_edges<<<dim3((NE + 255) / 256), dim3(256), 0, stream>>>(dst, etype, cnt, NE);
  make_inv<<<dim3((int)((nseg + 255) / 256)), dim3(256), 0, stream>>>(cnt, inv, nseg);
  xpose_bf16<<<dim3(512), dim3(256), 0, stream>>>(W1, W1t, NR * DIN, DHID);   // 1024x128
  xpose_bf16<<<dim3(64),  dim3(256), 0, stream>>>(root1, r1t, DIN, DHID);     // 128x128
  xpose_bf16<<<dim3(256), dim3(256), 0, stream>>>(W2, W2t, NR * DHID, DOUT);  // 1024x64
  xpose_bf16<<<dim3(32),  dim3(256), 0, stream>>>(root2, r2t, DHID, DOUT);    // 128x64

  // ---- layer 1: M = scatter_mean(x); h = relu(M@W1 + x@root1 + b1) ----
  scatter_mean_pk<<<dim3((NE * 32 + 255) / 256), dim3(256), 0, stream>>>(
      x, src, dst, etype, inv, M, NE);
  rgcn_gemm_wmma<DHID / 16, NR * DIN, DIN><<<dim3(NN / 16), dim3(32), 0, stream>>>(
      M, x, W1t, r1t, b1, h, 1);

  // ---- layer 2: M = scatter_mean(h); out = M@W2 + h@root2 + b2 ----
  zero_u128<<<dim3(4096), dim3(256), 0, stream>>>((uint4*)M, nM4);
  scatter_mean_pk<<<dim3((NE * 32 + 255) / 256), dim3(256), 0, stream>>>(
      h, src, dst, etype, inv, M, NE);
  rgcn_gemm_wmma<DOUT / 16, NR * DHID, DHID><<<dim3(NN / 16), dim3(32), 0, stream>>>(
      M, h, W2t, r2t, b2, out, 0);
}
